// BertSelfAttention_82154134438679
// MI455X (gfx1250) — matrix operand order repaired and checked by reading, not compiled, not run on hardware
//
#include <hip/hip_runtime.h>

// ---------- shapes ----------
#define B_   2
#define S_   2048
#define HID_ 1024
#define NH_  16
#define HD_  64
#define SKV_ 2048
#define LTOT 4096            // SKV_ + S_
#define BH_  (B_*NH_)        // 32
#define M_   (B_*S_)         // 4096 rows for QKV GEMM

typedef unsigned int   u32;
typedef unsigned short u16;
typedef __attribute__((ext_vector_type(16))) __bf16 v16bf;
typedef __attribute__((ext_vector_type(8)))  float  v8f;
typedef __attribute__((ext_vector_type(4)))  u32    u32x4;

union Frag {                 // 32 bytes = 8 VGPRs = one 16-bit WMMA A/B fragment
    u32x4 u[2];
    u32   w[8];
    v16bf v;
};
static_assert(sizeof(Frag) == 32, "frag size");

__device__ __forceinline__ u16 f2bf(float f) {
    u32 u = __float_as_uint(f);
    u32 r = u + 0x7FFFu + ((u >> 16) & 1u);   // round-to-nearest-even
    return (u16)(r >> 16);
}

__device__ __forceinline__ v8f wmma_bf16(const Frag& a, const Frag& b, v8f c) {
    return __builtin_amdgcn_wmma_f32_16x16x32_bf16(
        /*neg_a=*/false, a.v, /*neg_b=*/false, b.v,
        /*c_mod=*/(short)0, c, /*reuse_a=*/false, /*reuse_b=*/false);
}

__device__ __forceinline__ float lane_bcast(float x, int srcLane) {
    return __int_as_float(__builtin_amdgcn_ds_bpermute(srcLane << 2, __float_as_int(x)));
}

// ---------- kernel 1: fp32 -> bf16 elementwise ----------
__global__ void cvt_bf16(const float* __restrict__ src, u16* __restrict__ dst, int n) {
    int i = blockIdx.x * blockDim.x + threadIdx.x;
    int stride = gridDim.x * blockDim.x;
    for (; i < n; i += stride) dst[i] = f2bf(src[i]);
}

// ---------- kernel 2: KV cache -> bf16 K [BH][LTOT][HD] (first SKV rows)
// ----------                      and bf16 V^T [BH][HD][LTOT] (first SKV cols)
__global__ void cvt_cache(const float* __restrict__ kvs, const float* __restrict__ kvw,
                          u16* __restrict__ Kc, u16* __restrict__ VcT) {
    const int n = BH_ * SKV_ * HD_;           // 4,194,304 per tensor
    const float w = kvw[0];
    int i = blockIdx.x * blockDim.x + threadIdx.x;
    int stride = gridDim.x * blockDim.x;
    for (; i < n; i += stride) {
        int d   = i & (HD_ - 1);
        int skv = (i >> 6) & (SKV_ - 1);
        int bh  = i >> 17;                    // i / (HD_*SKV_)
        Kc [((size_t)bh * LTOT + skv) * HD_ + d]  = f2bf(w * kvs[i]);
        VcT[((size_t)bh * HD_  + d) * LTOT + skv] = f2bf(w * kvs[n + i]);
    }
}

// ---------- kernel 3: QKV projection  y = x @ W^T + b  (bf16 WMMA, f32 acc)
// grid (M/256, HID/64, 3), block 256 (8 waves). Wave -> 32x64 strip.
// A fragments ping-pong across K-steps; B fragments staggered inside a step.
__global__ void
__attribute__((amdgpu_flat_work_group_size(256, 256), amdgpu_waves_per_eu(1, 2)))
qkv_gemm(const u16* __restrict__ Xb,
         const u16* __restrict__ Wqb, const u16* __restrict__ Wkb,
         const u16* __restrict__ Wvb,
         const float* __restrict__ bq, const float* __restrict__ bk,
         const float* __restrict__ bv,
         u16* __restrict__ Qb, u16* __restrict__ Kc, u16* __restrict__ VcT) {
    const int which = blockIdx.z;
    const u16*   W    = (which == 0) ? Wqb : (which == 1) ? Wkb : Wvb;
    const float* bias = (which == 0) ? bq  : (which == 1) ? bk  : bv;

    const int wave = threadIdx.x >> 5;
    const int lane = threadIdx.x & 31;
    const int half = lane >> 4;
    const int ln   = lane & 15;

    const int mBase = blockIdx.x * 256 + wave * 32;   // 32-row strip of this wave
    const int nBase = blockIdx.y * 64;

    v8f acc[2][4];
#pragma unroll
    for (int mi = 0; mi < 2; ++mi)
#pragma unroll
        for (int j = 0; j < 4; ++j) acc[mi][j] = v8f{};

    const u16* xrow0 = Xb + (size_t)(mBase + ln) * HID_;
    const u16* xrow1 = xrow0 + 16 * HID_;
    const u16* wbase = W + (size_t)(nBase + ln) * HID_;

    auto loadA = [&](int k, Frag* a) {
        const u16* ap0 = xrow0 + k + half * 8;
        a[0].u[0] = *(const u32x4*)(ap0);
        a[0].u[1] = *(const u32x4*)(ap0 + 16);
        const u16* ap1 = xrow1 + k + half * 8;
        a[1].u[0] = *(const u32x4*)(ap1);
        a[1].u[1] = *(const u32x4*)(ap1 + 16);
    };
    auto loadB = [&](int k, int j, Frag& bf) {
        const u16* wrow = wbase + (size_t)j * 16 * HID_ + k + half * 8;
        bf.u[0] = *(const u32x4*)(wrow);
        bf.u[1] = *(const u32x4*)(wrow + 16);
    };
    // One 32-deep K step: 8 WMMAs; B[j+2] loads issue under B[j]'s WMMAs.
    auto kstep = [&](int k, const Frag* a) {
        Frag b[4];
        loadB(k, 0, b[0]);
        loadB(k, 1, b[1]);
#pragma unroll
        for (int j = 0; j < 4; ++j) {
            if (j < 2) loadB(k, j + 2, b[j + 2]);
            acc[0][j] = wmma_bf16(a[0], b[j], acc[0][j]);
            acc[1][j] = wmma_bf16(a[1], b[j], acc[1][j]);
        }
    };

    Frag aA[2], aB[2];
    loadA(0, aA);                                     // prologue
    for (int k = 0; k < HID_; k += 64) {
        loadA(k + 32, aB);                            // in flight during step A
        kstep(k, aA);
        if (k + 64 < HID_) loadA(k + 64, aA);         // in flight during step B
        kstep(k + 32, aB);
    }

    // epilogue: +bias, Q pre-scaled by d^-1/2; scatter into attention layouts
#pragma unroll
    for (int mi = 0; mi < 2; ++mi) {
#pragma unroll
        for (int j = 0; j < 4; ++j) {
            const int n = nBase + j * 16 + ln;         // C-frag: lane -> N
            const int h = n >> 6, d = n & 63;
            const float bb = bias[n];
#pragma unroll
            for (int v = 0; v < 8; ++v) {              // C-frag: VGPR -> M
                const int mr = mBase + mi * 16 + half * 8 + v;
                const int bt = mr >> 11;               // batch
                const int sr = mr & (S_ - 1);          // seq row
                float val = acc[mi][j][v] + bb;
                if (which == 0) {
                    val *= 0.125f;                     // q * HD^-0.5
                    Qb[((size_t)(bt * NH_ + h) * S_ + sr) * HD_ + d] = f2bf(val);
                } else if (which == 1) {
                    Kc[((size_t)(bt * NH_ + h) * LTOT + SKV_ + sr) * HD_ + d] = f2bf(val);
                } else {
                    VcT[((size_t)(bt * NH_ + h) * HD_ + d) * LTOT + SKV_ + sr] = f2bf(val);
                }
            }
        }
    }
}

// ---------- kernel 4: streaming-softmax attention, transposed score tiles
// grid (S/128, BH), block 256 (8 waves). Wave -> 16 query rows, L loop
// software-pipelined in 64-wide double chunks with K/V prefetch.
__global__ void
__attribute__((amdgpu_flat_work_group_size(256, 256), amdgpu_waves_per_eu(1, 2)))
attn(const u16* __restrict__ Qb, const u16* __restrict__ Kc,
     const u16* __restrict__ VcT, float* __restrict__ out) {
    const int bh = blockIdx.y;
    const int bt = bh >> 4, h = bh & 15;
    const int wave = threadIdx.x >> 5;
    const int lane = threadIdx.x & 31;
    const int half = lane >> 4;
    const int ln   = lane & 15;

    const int q0 = blockIdx.x * 128 + wave * 16;
    const u16* Qt = Qb  + ((size_t)bh * S_ + q0) * HD_;
    const u16* Kh = Kc  + (size_t)bh * LTOT * HD_;
    const u16* Vh = VcT + (size_t)bh * HD_ * LTOT;

    // Q as B-fragments of T = K x Q^T  (loop invariant): lane -> q, VGPR -> d
    Frag qf0, qf1;
    {
        const u16* qrow = Qt + (size_t)ln * HD_;
        qf0.u[0] = *(const u32x4*)(qrow + half * 8);
        qf0.u[1] = *(const u32x4*)(qrow + half * 8 + 16);
        qf1.u[0] = *(const u32x4*)(qrow + 32 + half * 8);
        qf1.u[1] = *(const u32x4*)(qrow + 32 + half * 8 + 16);
    }

    float mrun = -INFINITY, srun = 0.f;
    v8f O[4] = {v8f{}, v8f{}, v8f{}, v8f{}};       // ctx[16 q][64 d], lane->d, VGPR->q

    // K fragments for a 32-row L chunk: [tile0 d0..31, tile0 d32..63, tile1 ...]
    auto loadK = [&](Frag* kf, int l0) {
        const u16* kr0 = Kh + (size_t)(l0 + ln) * HD_;
        kf[0].u[0] = *(const u32x4*)(kr0 + half * 8);
        kf[0].u[1] = *(const u32x4*)(kr0 + 16 + half * 8);
        kf[1].u[0] = *(const u32x4*)(kr0 + 32 + half * 8);
        kf[1].u[1] = *(const u32x4*)(kr0 + 48 + half * 8);
        const u16* kr1 = kr0 + 16 * HD_;
        kf[2].u[0] = *(const u32x4*)(kr1 + half * 8);
        kf[2].u[1] = *(const u32x4*)(kr1 + 16 + half * 8);
        kf[3].u[0] = *(const u32x4*)(kr1 + 32 + half * 8);
        kf[3].u[1] = *(const u32x4*)(kr1 + 48 + half * 8);
    };
    auto loadV = [&](Frag* vf, int l0, int jd0) {
#pragma unroll
        for (int jd = jd0; jd < jd0 + 2; ++jd) {
            const u16* vr = Vh + (size_t)(jd * 16 + ln) * LTOT + l0 + half * 8;
            vf[jd].u[0] = *(const u32x4*)(vr);
            vf[jd].u[1] = *(const u32x4*)(vr + 16);
        }
    };

    auto chunk = [&](int l0, const Frag* kf) {
        Frag vf[4];
        loadV(vf, l0, 0);                              // hide under score WMMAs
        // streaming prefetch three chunks ahead (K rows + V^T rows)
        if (l0 + 96 < LTOT) {
            __builtin_prefetch(Kh + (size_t)(l0 + 96 + ln) * HD_, 0, 0);
            __builtin_prefetch(Vh + (size_t)(((lane & 3) * 16) + (lane >> 2)) * LTOT + l0 + 96, 0, 0);
        }

        // transposed scores: T[l][q], two 16x16 tiles (l0..15, l16..31)
        v8f T0 = v8f{}, T1 = v8f{};
        T0 = wmma_bf16(kf[0], qf0, T0);
        T0 = wmma_bf16(kf[1], qf1, T0);
        T1 = wmma_bf16(kf[2], qf0, T1);
        T1 = wmma_bf16(kf[3], qf1, T1);
        loadV(vf, l0, 2);                              // hide under softmax VALU

        // per-lane (fixed q = ln) streaming softmax over this chunk's l subset
        float p0[8], p1[8];
        float cmax = -INFINITY;
#pragma unroll
        for (int v = 0; v < 8; ++v) {
            p0[v] = T0[v] * 0.125f;                 // second 1/sqrt(d) scale
            p1[v] = T1[v] * 0.125f;
            cmax = fmaxf(cmax, fmaxf(p0[v], p1[v]));
        }
        cmax = fmaxf(cmax, lane_bcast(cmax, lane ^ 16));   // combine half-wave partners
        const float mnew = fmaxf(mrun, cmax);
        const float corr = __expf(mrun - mnew);
        float ssum = 0.f;
#pragma unroll
        for (int v = 0; v < 8; ++v) {
            p0[v] = __expf(p0[v] - mnew);
            p1[v] = __expf(p1[v] - mnew);
            ssum += p0[v] + p1[v];
        }
        srun = srun * corr + ssum;
        mrun = mnew;

        // rescale O: factor indexed by q = 8*half + v (broadcast from q-lanes)
        float fv[8];
        const int bl = half * 8;
#pragma unroll
        for (int v = 0; v < 8; ++v) fv[v] = lane_bcast(corr, bl + v);
#pragma unroll
        for (int jd = 0; jd < 4; ++jd)
#pragma unroll
            for (int v = 0; v < 8; ++v) O[jd][v] *= fv[v];

        // P A-fragment: pure per-lane repack of T-layout probabilities
        Frag pf;
#pragma unroll
        for (int j = 0; j < 4; ++j) {
            pf.w[j]     = (u32)f2bf(p0[2 * j]) | ((u32)f2bf(p0[2 * j + 1]) << 16);
            pf.w[4 + j] = (u32)f2bf(p1[2 * j]) | ((u32)f2bf(p1[2 * j + 1]) << 16);
        }

        // ctx += P x V  (V^T rows are contiguous in l)
#pragma unroll
        for (int jd = 0; jd < 4; ++jd) O[jd] = wmma_bf16(pf, vf[jd], O[jd]);
    };

    // ping-pong over 64-wide double chunks: K loads stay one chunk ahead
    Frag kA[4], kB[4];
    loadK(kA, 0);
    for (int l0 = 0; l0 < LTOT; l0 += 64) {
        loadK(kB, l0 + 32);
        chunk(l0, kA);
        if (l0 + 64 < LTOT) loadK(kA, l0 + 64);
        chunk(l0 + 32, kB);
    }

    // finalize: 1/rowsum, write ctx as [B][S][NH*HD]
    const float stot = srun + lane_bcast(srun, lane ^ 16);
    const float inv = 1.0f / stot;
    float fin[8];
    const int bl = half * 8;
#pragma unroll
    for (int v = 0; v < 8; ++v) fin[v] = lane_bcast(inv, bl + v);

#pragma unroll
    for (int jd = 0; jd < 4; ++jd) {
        const int col = h * HD_ + jd * 16 + ln;
#pragma unroll
        for (int v = 0; v < 8; ++v) {
            const int qr = q0 + half * 8 + v;
            out[((size_t)bt * S_ + qr) * HID_ + col] = O[jd][v] * fin[v];
        }
    }
}

// ---------- host launcher ----------
extern "C" void kernel_launch(void* const* d_in, const int* in_sizes, int n_in,
                              void* d_out, int out_size, void* d_ws, size_t ws_size,
                              hipStream_t stream) {
    const float* hs  = (const float*)d_in[0];
    const float* kvs = (const float*)d_in[1];
    const float* Wq  = (const float*)d_in[2];
    const float* bq  = (const float*)d_in[3];
    const float* Wk  = (const float*)d_in[4];
    const float* bk  = (const float*)d_in[5];
    const float* Wv  = (const float*)d_in[6];
    const float* bv  = (const float*)d_in[7];
    const float* kvw = (const float*)d_in[8];

    char* ws = (char*)d_ws;
    // bf16 workspace layout (54 MB total; whole pipeline fits in 192 MB L2)
    u16* Xb  = (u16*)(ws + 0);             // [4096][1024]            8 MB
    u16* Wqb = (u16*)(ws + 8388608);       // [1024][1024]            2 MB
    u16* Wkb = (u16*)(ws + 10485760);      //                         2 MB
    u16* Wvb = (u16*)(ws + 12582912);      //                         2 MB
    u16* Qb  = (u16*)(ws + 14680064);      // [BH][2048][64]          8 MB
    u16* Kc  = (u16*)(ws + 23068672);      // [BH][4096][64]         16 MB
    u16* VcT = (u16*)(ws + 39845888);      // [BH][64][4096] (V^T)   16 MB

    cvt_bf16<<<1024, 256, 0, stream>>>(hs, Xb, M_ * HID_);
    cvt_bf16<<<512,  256, 0, stream>>>(Wq, Wqb, HID_ * HID_);
    cvt_bf16<<<512,  256, 0, stream>>>(Wk, Wkb, HID_ * HID_);
    cvt_bf16<<<512,  256, 0, stream>>>(Wv, Wvb, HID_ * HID_);
    cvt_cache<<<1024, 256, 0, stream>>>(kvs, kvw, Kc, VcT);

    qkv_gemm<<<dim3(M_ / 256, HID_ / 64, 3), 256, 0, stream>>>(
        Xb, Wqb, Wkb, Wvb, bq, bk, bv, Qb, Kc, VcT);

    attn<<<dim3(S_ / 128, BH_), 256, 0, stream>>>(Qb, Kc, VcT, (float*)d_out);
}
